// GRUCellv2_86251533238683
// MI455X (gfx1250) — matrix operand order repaired and checked by reading, not compiled, not run on hardware
//
#include <hip/hip_runtime.h>
#include <cstdint>
#include <cstddef>

typedef __attribute__((ext_vector_type(16))) _Float16     v16h;
typedef __attribute__((ext_vector_type(2)))  __fp16       v2fp;
typedef __attribute__((ext_vector_type(8)))  float        v8f;
typedef __attribute__((ext_vector_type(4)))  float        v4f;
typedef __attribute__((ext_vector_type(4)))  unsigned int v4u;

#define B_SZ 8192
#define HD   1024
#define KT   32      // K per WMMA step
#define KPAD 40      // LDS row pitch in halves (80B = 20 banks -> conflict free)
#define MT   64      // block tile M (2 waves x 32)
#define NT   128     // block tile N (4 waves x 32)
// packed weight layout: halves, fragment-native:
//   idx = (((t*32 + kb)*64 + nb)*32 + lane)*16 + j
// t: gate-source tile 0..5 (wx0,wx1,wx2,wh0,wh1,wh2), kb: K/32, nb: N/16
#define T_STRIDE (32u * 64u * 512u)   // halves per tile t  (=1048576, 2MB)

__device__ __forceinline__ unsigned int pack2(float a, float b) {
    union { v2fp h; unsigned int u; } c;
    c.h = __builtin_amdgcn_cvt_pkrtz(a, b);   // v_cvt_pk_rtz_f16_f32
    return c.u;
}

__device__ __forceinline__ v16h ld_frag_lds(const _Float16* p0, const _Float16* p1) {
    union { v16h h; v4u u[2]; } f;
    f.u[0] = *(const v4u*)p0;   // ds_load_b128
    f.u[1] = *(const v4u*)p1;   // ds_load_b128
    return f.h;
}

__device__ __forceinline__ v16h ld_frag_g(const _Float16* p) {
    union { v16h h; v4u u[2]; } f;
    f.u[0] = *(const v4u*)p;         // global_load_b128, fully coalesced
    f.u[1] = *(const v4u*)(p + 8);   // global_load_b128
    return f.h;
}

__device__ __forceinline__ float fast_sigmoid(float x) {
    return 1.0f / (1.0f + __expf(-x));
}
__device__ __forceinline__ float fast_tanh(float x) {
    return 1.0f - 2.0f / (__expf(2.0f * x) + 1.0f);
}

#define WMMA(A, Bv, C) __builtin_amdgcn_wmma_f32_16x16x32_f16(false, A, false, Bv, (short)0, C, false, false)

// ---------------------------------------------------------------------------
// Prep: convert wx/wh f32 -> f16, pre-swizzled into per-lane B-fragment layout
// One 16B chunk per thread: 786432 chunks total (12MB out, 24MB in).
// ---------------------------------------------------------------------------
__global__ __launch_bounds__(256)
void prep_weights(const float* __restrict__ wx, const float* __restrict__ wh,
                  _Float16* __restrict__ w2) {
    const int id   = blockIdx.x * 256 + threadIdx.x;  // chunk index
    const int c    = id & 1;           // which 8-half chunk of the lane's 16
    const int lane = (id >> 1) & 31;
    const int nb   = (id >> 6) & 63;
    const int kb   = (id >> 12) & 31;
    const int t    = id >> 17;         // 0..5
    const float* src = (t < 3) ? (wx + (size_t)t * HD * HD)
                               : (wh + (size_t)(t - 3) * HD * HD);
    // B-fragment k order used here (contiguous, matches A): lanes 0-15 -> N=nb*16+lane, K 0..15 of the slice;
    //                       lanes 16-31 -> same N set, K 16..31.
    const int n  = nb * 16 + (lane & 15);
    const int k0 = kb * 32 + ((lane & 16) ? 16 : 0) + c * 8;
    float f[8];
#pragma unroll
    for (int j = 0; j < 8; ++j) f[j] = src[(size_t)(k0 + j) * HD + n];
    v4u p;
    p.x = pack2(f[0], f[1]);  p.y = pack2(f[2], f[3]);
    p.z = pack2(f[4], f[5]);  p.w = pack2(f[6], f[7]);
    *(v4u*)(w2 + (size_t)id * 8) = p;  // global_store_b128
}

// ---------------------------------------------------------------------------
// Fused GRU: 32x32 per wave, 64x128 per block, K-step 32.
// A (x,hid) staged through LDS with packed cvt; B straight from L2-resident w2.
// ---------------------------------------------------------------------------
__global__ __launch_bounds__(256)
void gru_wmma_kernel(const float* __restrict__ x,  const float* __restrict__ hid,
                     const _Float16* __restrict__ w2,
                     const float* __restrict__ bx, const float* __restrict__ bh,
                     float* __restrict__ out) {
    __shared__ _Float16 Ax[MT * KPAD];   // 5120 B
    __shared__ _Float16 Ah[MT * KPAD];   // 5120 B

    const int tid   = threadIdx.x;
    const int lane  = tid & 31;
    const int wave  = tid >> 5;
    const int waveM = wave & 1;     // 2 waves along M (32 rows each)
    const int waveN = wave >> 1;    // 4 waves along N (32 cols each)
    const int rowBase = blockIdx.x * MT;
    const int nbBase  = blockIdx.y * (NT / 16) + waveN * 2;  // N/16 fragment index

    v8f acc_r[2][2]  = {{}};   // [m_sub][n_sub]
    v8f acc_z[2][2]  = {{}};
    v8f acc_nx[2][2] = {{}};
    v8f acc_nh[2][2] = {{}};

    // A staging indices: thread -> (row, 8-wide k chunk)
    const int am = tid >> 2;
    const int ak = (tid & 3) * 8;
    // A fragment indices: lanes 16-31 hold K 16..31 (contiguous, matches B)
    const int akoff = (lane & 16) ? 16 : 0;
    // B fragment per-lane offset in halves
    const int blane = lane * 16;

    for (int kt = 0; kt < HD; kt += KT) {
        const int kb = kt >> 5;
        __syncthreads();
        // ---- stage x/hid tile: 2 x (2 global_load_b128 -> 4 cvt_pk -> ds_store_b128)
        {
            const size_t gro = (size_t)(rowBase + am) * HD + kt + ak;
            const v4f* px = (const v4f*)(x + gro);
            const v4f* ph = (const v4f*)(hid + gro);
            v4f a0 = px[0], a1 = px[1];
            v4f b0 = ph[0], b1 = ph[1];
            v4u pa, pb;
            pa.x = pack2(a0.x, a0.y);  pa.y = pack2(a0.z, a0.w);
            pa.z = pack2(a1.x, a1.y);  pa.w = pack2(a1.z, a1.w);
            pb.x = pack2(b0.x, b0.y);  pb.y = pack2(b0.z, b0.w);
            pb.z = pack2(b1.x, b1.y);  pb.w = pack2(b1.z, b1.w);
            *(v4u*)&Ax[am * KPAD + ak] = pa;
            *(v4u*)&Ah[am * KPAD + ak] = pb;
        }
        __syncthreads();

        // ---- A fragments: 2 M-subtiles x {x, hid}
        v16h axf[2], ahf[2];
#pragma unroll
        for (int m = 0; m < 2; ++m) {
            const int r = (waveM * 32 + m * 16 + (lane & 15)) * KPAD + akoff;
            axf[m] = ld_frag_lds(&Ax[r], &Ax[r + 8]);
            ahf[m] = ld_frag_lds(&Ah[r], &Ah[r + 8]);
        }

        // ---- 24 WMMA: each B fragment loaded once, used by both M-subtiles
#pragma unroll
        for (int ns = 0; ns < 2; ++ns) {
            const _Float16* wb =
                w2 + ((size_t)kb * 64 + (nbBase + ns)) * 512 + blane;
            v16h b;
            b = ld_frag_g(wb + 0u * T_STRIDE);                 // wx[r]
            acc_r[0][ns]  = WMMA(axf[0], b, acc_r[0][ns]);
            acc_r[1][ns]  = WMMA(axf[1], b, acc_r[1][ns]);
            b = ld_frag_g(wb + 3u * T_STRIDE);                 // wh[r]
            acc_r[0][ns]  = WMMA(ahf[0], b, acc_r[0][ns]);
            acc_r[1][ns]  = WMMA(ahf[1], b, acc_r[1][ns]);
            b = ld_frag_g(wb + 1u * T_STRIDE);                 // wx[z]
            acc_z[0][ns]  = WMMA(axf[0], b, acc_z[0][ns]);
            acc_z[1][ns]  = WMMA(axf[1], b, acc_z[1][ns]);
            b = ld_frag_g(wb + 4u * T_STRIDE);                 // wh[z]
            acc_z[0][ns]  = WMMA(ahf[0], b, acc_z[0][ns]);
            acc_z[1][ns]  = WMMA(ahf[1], b, acc_z[1][ns]);
            b = ld_frag_g(wb + 2u * T_STRIDE);                 // wx[n]
            acc_nx[0][ns] = WMMA(axf[0], b, acc_nx[0][ns]);
            acc_nx[1][ns] = WMMA(axf[1], b, acc_nx[1][ns]);
            b = ld_frag_g(wb + 5u * T_STRIDE);                 // wh[n]
            acc_nh[0][ns] = WMMA(ahf[0], b, acc_nh[0][ns]);
            acc_nh[1][ns] = WMMA(ahf[1], b, acc_nh[1][ns]);
        }
    }

    // ---- fused epilogue: bias + sigmoid/tanh + blend
    // C/D layout: VGPR v -> row M = v (lanes 0-15) / v+8 (lanes 16-31); lane%16 = N
    const int mBase = rowBase + waveM * 32 + ((lane & 16) ? 8 : 0);
#pragma unroll
    for (int ns = 0; ns < 2; ++ns) {
        const int h   = blockIdx.y * NT + waveN * 32 + ns * 16 + (lane & 15);
        const float br  = bx[h]          + bh[h];
        const float bz  = bx[HD + h]     + bh[HD + h];
        const float bnx = bx[2 * HD + h];
        const float bnh = bh[2 * HD + h];
#pragma unroll
        for (int m = 0; m < 2; ++m) {
#pragma unroll
            for (int v = 0; v < 8; ++v) {
                const int   mm = mBase + m * 16 + v;
                const float hv = hid[(size_t)mm * HD + h];
                const float r  = fast_sigmoid(acc_r[m][ns][v] + br);
                const float z  = fast_sigmoid(acc_z[m][ns][v] + bz);
                const float n  = fast_tanh((acc_nx[m][ns][v] + bnx) +
                                           r * (acc_nh[m][ns][v] + bnh));
                out[(size_t)mm * HD + h] = (1.0f - z) * n + z * hv;
            }
        }
    }
}

extern "C" void kernel_launch(void* const* d_in, const int* in_sizes, int n_in,
                              void* d_out, int out_size, void* d_ws, size_t ws_size,
                              hipStream_t stream) {
    (void)in_sizes; (void)n_in; (void)out_size; (void)ws_size;
    const float* x   = (const float*)d_in[0];
    const float* hid = (const float*)d_in[1];
    const float* wx  = (const float*)d_in[2];
    const float* wh  = (const float*)d_in[3];
    const float* bx  = (const float*)d_in[4];
    const float* bh  = (const float*)d_in[5];
    float* out = (float*)d_out;
    _Float16* w2 = (_Float16*)d_ws;   // needs 6*1024*1024*2 = 12 MB scratch

    // 1) repack weights into fragment-native f16 (one-shot, ~1.5us of HBM time)
    prep_weights<<<(6 * 32 * 64 * 64) / 256, 256, 0, stream>>>(wx, wh, w2);
    // 2) fused GRU GEMM + epilogue
    dim3 grid(B_SZ / MT, HD / NT);    // 128 x 8 = 1024 blocks
    gru_wmma_kernel<<<grid, 256, 0, stream>>>(x, hid, w2, bx, bh, out);
}
